// CrossAttention_67233418051751
// MI455X (gfx1250) — matrix operand order repaired and checked by reading, not compiled, not run on hardware
//
#include <hip/hip_runtime.h>
#include <hip/hip_bf16.h>

// ---------------------------------------------------------------------------
// CrossAttention for MI455X (gfx1250, wave32, WMMA + Tensor Data Mover).
// Pipeline: cvt->bf16, GEMM(Q), GEMM(KV), LN+RoPE(Q), LN(K)+transpose(V),
// fused flash attention (TDM-staged K/V, WMMA QK^T + online softmax + WMMA PV),
// GEMM(out). All matmuls: v_wmma_f32_16x16x32_bf16, f32 accumulate.
// GEMM/attention operand tiles are DMA'd into LDS by the TDM
// (tensor_load_to_lds, TENSORcnt) with double buffering.
// ---------------------------------------------------------------------------

typedef __attribute__((ext_vector_type(16))) __bf16 v16bf;
typedef __attribute__((ext_vector_type(8)))  float  v8f;
typedef __attribute__((ext_vector_type(4)))  unsigned int u32x4;
typedef __attribute__((ext_vector_type(8)))  int    i32x8;
typedef __attribute__((ext_vector_type(4)))  int    i32x4;

union Frag16 {
    uint4  q[2];   // 2 x 128-bit loads
    v16bf  v;      // 16 bf16 = 8 VGPRs (wave32 WMMA 16-bit operand)
};

// Problem constants
#define BATCH 2
#define S1    2048
#define S2    1024
#define NHEAD 16
#define HDIM  64
#define CDIM  1024

// ---------------------------------------------------------------------------
// LDS byte offset of a __shared__ object: flat LDS-aperture address keeps the
// LDS offset in addr[31:0] (ISA 10.2 aperture mapping), so truncation works.
// ---------------------------------------------------------------------------
__device__ __forceinline__ unsigned int lds_off(const void* p) {
    return (unsigned int)(unsigned long long)p;
}

// ---------------------------------------------------------------------------
// TDM 2D tile load: global (bf16, row-major, row stride = stride0 elements)
// -> LDS (packed row-major tile_d1 x tile_d0). D# packed per CDNA5 ISA 8.3/8.4:
//  g0: count=1 | lds_addr | global_addr[56:0] | type=2
//  g1: data_size=1(2B) | tensor_dim0/1 | tile_dim0/1 | tensor_dim0_stride
// Groups 2/3 zero (2D tensor). Issue from ONE wave only (EXEC ignored by TDM).
// Toolchain here uses the 6-arg builtin: (g0, g1, g2, g3, g_ext, cpol).
// ---------------------------------------------------------------------------
__device__ __forceinline__ void tdm_load_2d(unsigned int lds_byte,
                                            const void* gptr,
                                            unsigned int tdim0, unsigned int tdim1,
                                            unsigned int tile0, unsigned int tile1,
                                            unsigned long long stride0) {
    const unsigned long long ga = (unsigned long long)gptr;
    u32x4 g0;
    g0.x = 1u;                                                  // count=1
    g0.y = lds_byte;                                            // lds_addr
    g0.z = (unsigned int)ga;                                    // gaddr[31:0]
    g0.w = (unsigned int)((ga >> 32) & 0x1FFFFFFu) | (2u << 30);// gaddr[56:32]|type=2
    i32x8 g1;
    g1[0] = (int)(1u << 16);                                    // data_size=1 (2B)
    g1[1] = (int)((tdim0 & 0xFFFFu) << 16);                     // tensor_dim0 lo
    g1[2] = (int)((tdim0 >> 16) | ((tdim1 & 0xFFFFu) << 16));   // dim0 hi|dim1 lo
    g1[3] = (int)((tdim1 >> 16) | (tile0 << 16));               // dim1 hi|tile_dim0
    g1[4] = (int)(tile1 & 0xFFFFu);                             // tile_dim1, tile_dim2=0
    g1[5] = (int)(unsigned int)(stride0 & 0xFFFFFFFFu);         // dim0_stride lo
    g1[6] = (int)(unsigned int)((stride0 >> 32) & 0xFFFFu);     // dim0_stride hi
    g1[7] = 0;
    i32x4 z4 = {0, 0, 0, 0};
    i32x8 z8 = {0, 0, 0, 0, 0, 0, 0, 0};
    __builtin_amdgcn_tensor_load_to_lds(g0, g1, z4, z4, z8, 0);
}

// ---------------------------------------------------------------------------
// f32 -> bf16 conversion
// ---------------------------------------------------------------------------
__global__ void cvt_f32_bf16(const float* __restrict__ src,
                             __bf16* __restrict__ dst, int n) {
    int i = blockIdx.x * blockDim.x + threadIdx.x;
    int stride = gridDim.x * blockDim.x;
    for (; i < n; i += stride) dst[i] = (__bf16)src[i];
}

// ---------------------------------------------------------------------------
// GEMM: C[M,N] = A[M,K](bf16 rm) * W[N,K](bf16 rm)^T + bias, f32 out.
// Block = 128 threads = 4 waves; block tile 128x128; wave tile 64x64.
// A/B 128x32 slabs staged to LDS by the TDM, double buffered:
//   wave0: tensor_load_to_lds (cur ready? s_wait_tensorcnt) -> barrier ->
//   issue next slab -> all waves compute from LDS (ds_load_b128 fragments).
// Fragment maps follow the documented 16-bit WMMA VGPR layouts.
// ---------------------------------------------------------------------------
__global__ __launch_bounds__(128)
void gemm_bf16_wmma(const __bf16* __restrict__ A, const __bf16* __restrict__ W,
                    const float* __restrict__ bias, float* __restrict__ C,
                    int M, int N, int K) {
    __shared__ __bf16 Asm[2][128 * 32];   // 8KB each
    __shared__ __bf16 Bsm[2][128 * 32];   // 8KB each

    const int w    = threadIdx.x >> 5;
    const int lane = threadIdx.x & 31;
    const int g    = lane >> 4;
    const int l15  = lane & 15;

    const int rblk = blockIdx.y * 128;
    const int nblk = blockIdx.x * 128;
    const int rw   = (w >> 1) * 64;       // wave row offset within block tile
    const int nw   = (w & 1) * 64;        // wave col offset within block tile

    v8f acc[4][4] = {};

    if (threadIdx.x < 32) {               // wave 0 drives the TDM
        tdm_load_2d(lds_off(Asm[0]), A + (size_t)rblk * K,
                    (unsigned)K, (unsigned)(M - rblk), 32u, 128u,
                    (unsigned long long)K);
        tdm_load_2d(lds_off(Bsm[0]), W + (size_t)nblk * K,
                    (unsigned)K, (unsigned)(N - nblk), 32u, 128u,
                    (unsigned long long)K);
    }

    for (int k0 = 0; k0 < K; k0 += 32) {
        const int cur = (k0 >> 5) & 1;
        if (threadIdx.x < 32) __builtin_amdgcn_s_wait_tensorcnt(0);
        __syncthreads();                  // slab `cur` visible to all waves
        if (k0 + 32 < K && threadIdx.x < 32) {
            tdm_load_2d(lds_off(Asm[cur ^ 1]), A + (size_t)rblk * K + k0 + 32,
                        (unsigned)(K - k0 - 32), (unsigned)(M - rblk), 32u, 128u,
                        (unsigned long long)K);
            tdm_load_2d(lds_off(Bsm[cur ^ 1]), W + (size_t)nblk * K + k0 + 32,
                        (unsigned)(K - k0 - 32), (unsigned)(N - nblk), 32u, 128u,
                        (unsigned long long)K);
        }

        Frag16 a[4], b[4];
#pragma unroll
        for (int mt = 0; mt < 4; ++mt) {
            const __bf16* p = &Asm[cur][(rw + mt * 16 + l15) * 32 + g * 8];
            a[mt].q[0] = *(const uint4*)(p);
            a[mt].q[1] = *(const uint4*)(p + 16);
        }
#pragma unroll
        for (int nt = 0; nt < 4; ++nt) {
            const __bf16* p = &Bsm[cur][(nw + nt * 16 + l15) * 32 + g * 8];
            b[nt].q[0] = *(const uint4*)(p);
            b[nt].q[1] = *(const uint4*)(p + 16);
        }
#pragma unroll
        for (int mt = 0; mt < 4; ++mt)
#pragma unroll
            for (int nt = 0; nt < 4; ++nt)
                acc[mt][nt] = __builtin_amdgcn_wmma_f32_16x16x32_bf16(
                    false, a[mt].v, false, b[nt].v, (short)0, acc[mt][nt],
                    false, false);

        asm volatile("s_wait_dscnt 0" ::: "memory");  // WAR vs next TDM write
        __syncthreads();
    }

#pragma unroll
    for (int mt = 0; mt < 4; ++mt) {
#pragma unroll
        for (int nt = 0; nt < 4; ++nt) {
            const int col = nblk + nw + nt * 16 + l15;
            const float bv = bias[col];
#pragma unroll
            for (int v = 0; v < 8; ++v) {
                const int row = rblk + rw + mt * 16 + v + 8 * g;
                C[(size_t)row * N + col] = acc[mt][nt][v] + bv;
            }
        }
    }
}

// ---------------------------------------------------------------------------
// LayerNorm(head_dim) + RoPE + scale on Q; write bf16 in (b,H,s1,hd) layout.
// One wave per (row, head); lane handles the RoPE pair (2*lane, 2*lane+1).
// ---------------------------------------------------------------------------
__global__ __launch_bounds__(128)
void ln_rope_q(const float* __restrict__ Qf, const float* __restrict__ w,
               const float* __restrict__ b, __bf16* __restrict__ Qb) {
    const int wid  = (blockIdx.x * blockDim.x + threadIdx.x) >> 5;
    const int lane = threadIdx.x & 31;
    const int h    = wid & (NHEAD - 1);
    const int row  = wid >> 4;              // row in [0, BATCH*S1)
    const int bpos = row / S1;
    const int s    = row - bpos * S1;

    const float* src = Qf + (size_t)row * CDIM + h * HDIM;
    float e0 = src[2 * lane];
    float e1 = src[2 * lane + 1];

    float sum = e0 + e1;
    float sq  = e0 * e0 + e1 * e1;
#pragma unroll
    for (int m = 16; m >= 1; m >>= 1) {
        sum += __shfl_xor(sum, m, 32);
        sq  += __shfl_xor(sq,  m, 32);
    }
    const float mu  = sum * (1.0f / HDIM);
    const float var = sq * (1.0f / HDIM) - mu * mu;
    const float rs  = rsqrtf(var + 1e-6f);

    const float n0 = (e0 - mu) * rs * w[2 * lane]     + b[2 * lane];
    const float n1 = (e1 - mu) * rs * w[2 * lane + 1] + b[2 * lane + 1];

    const float inv_freq = __powf(10000.0f, -(float)(2 * lane) * (1.0f / HDIM));
    const float ang = (float)s * inv_freq;
    float sn, cs;
    __sincosf(ang, &sn, &cs);
    const float scale = 0.125f;  // 64^-0.5
    const float o0 = (n0 * cs - n1 * sn) * scale;
    const float o1 = (n0 * sn + n1 * cs) * scale;

    __bf16* dst = Qb + ((size_t)(bpos * NHEAD + h) * S1 + s) * HDIM;
    dst[2 * lane]     = (__bf16)o0;
    dst[2 * lane + 1] = (__bf16)o1;
}

// ---------------------------------------------------------------------------
// LayerNorm(head_dim) on K -> bf16 (b,H,s2,hd); V -> bf16 transposed
// (b,H,hd,s2) so PV WMMA B-fragments are contiguous along s2.
// ---------------------------------------------------------------------------
__global__ __launch_bounds__(128)
void ln_k_tr_v(const float* __restrict__ KVf, const float* __restrict__ kw,
               const float* __restrict__ kb, __bf16* __restrict__ Kb,
               __bf16* __restrict__ Vt) {
    const int wid  = (blockIdx.x * blockDim.x + threadIdx.x) >> 5;
    const int lane = threadIdx.x & 31;
    const int h    = wid & (NHEAD - 1);
    const int row  = wid >> 4;              // row in [0, BATCH*S2)
    const int bpos = row / S2;
    const int s    = row - bpos * S2;

    const float* base = KVf + (size_t)row * (2 * CDIM);

    float e0 = base[h * HDIM + lane];
    float e1 = base[h * HDIM + lane + 32];
    float sum = e0 + e1;
    float sq  = e0 * e0 + e1 * e1;
#pragma unroll
    for (int m = 16; m >= 1; m >>= 1) {
        sum += __shfl_xor(sum, m, 32);
        sq  += __shfl_xor(sq,  m, 32);
    }
    const float mu  = sum * (1.0f / HDIM);
    const float var = sq * (1.0f / HDIM) - mu * mu;
    const float rs  = rsqrtf(var + 1e-6f);

    __bf16* kd = Kb + ((size_t)(bpos * NHEAD + h) * S2 + s) * HDIM;
    kd[lane]      = (__bf16)((e0 - mu) * rs * kw[lane]      + kb[lane]);
    kd[lane + 32] = (__bf16)((e1 - mu) * rs * kw[lane + 32] + kb[lane + 32]);

    const float v0 = base[CDIM + h * HDIM + lane];
    const float v1 = base[CDIM + h * HDIM + lane + 32];
    __bf16* vbase = Vt + (size_t)(bpos * NHEAD + h) * HDIM * S2;
    vbase[(size_t)lane * S2 + s]        = (__bf16)v0;
    vbase[(size_t)(lane + 32) * S2 + s] = (__bf16)v1;
}

// ---------------------------------------------------------------------------
// Fused attention. Per (b,h): block = 128 threads = 4 waves, 128 q-rows.
// K/V chunks (32 keys) are shared by all 4 waves -> TDM-staged into LDS with
// double buffering. Per wave: 32 q-rows; S = Q K^T (WMMA), online softmax
// (shfl_xor reductions over 16-lane row halves), P C-layout -> A-layout via
// per-wave LDS (ds_store_b16 / s_wait_dscnt / ds_load_b128), O += P V (WMMA).
// ---------------------------------------------------------------------------
__global__ __launch_bounds__(128)
void attention_wmma(const __bf16* __restrict__ Qb, const __bf16* __restrict__ Kb,
                    const __bf16* __restrict__ Vt, __bf16* __restrict__ Ob) {
    __shared__ __bf16 Ksm[2][32 * 64];     // 32 keys x 64 hd      (4KB each)
    __shared__ __bf16 Vsm[2][64 * 32];     // 64 hd   x 32 keys    (4KB each)
    __shared__ __bf16 ldsP[4 * 32 * 32];   // per-wave P transpose (2KB each)

    const int w    = threadIdx.x >> 5;
    const int lane = threadIdx.x & 31;
    const int g    = lane >> 4;
    const int l15  = lane & 15;

    const int bh   = blockIdx.y;          // b*H + h
    const int bpos = bh >> 4;
    const int h    = bh & 15;
    const int q0   = blockIdx.x * 128 + w * 32;

    const __bf16* Qbase = Qb + (size_t)bh * S1 * HDIM;
    const __bf16* Kbase = Kb + (size_t)bh * S2 * HDIM;
    const __bf16* Vbase = Vt + (size_t)bh * HDIM * S2;
    __bf16* lp = ldsP + w * (32 * 32);

    // Q A-fragments: 2 m-tiles x 2 d-steps, loop invariant (from global/L2)
    Frag16 aq[2][2];
#pragma unroll
    for (int mt = 0; mt < 2; ++mt)
#pragma unroll
        for (int d = 0; d < 2; ++d) {
            const __bf16* p =
                Qbase + (size_t)(q0 + mt * 16 + l15) * HDIM + d * 32 + g * 8;
            aq[mt][d].q[0] = *(const uint4*)(p);
            aq[mt][d].q[1] = *(const uint4*)(p + 16);
        }

    float mrun[2][8], lrun[2][8];
#pragma unroll
    for (int mt = 0; mt < 2; ++mt)
#pragma unroll
        for (int v = 0; v < 8; ++v) { mrun[mt][v] = -1e30f; lrun[mt][v] = 0.0f; }
    v8f oacc[2][4] = {};

    if (threadIdx.x < 32) {               // wave 0 drives the TDM
        tdm_load_2d(lds_off(Ksm[0]), Kbase, 64u, (unsigned)S2, 64u, 32u, 64ull);
        tdm_load_2d(lds_off(Vsm[0]), Vbase, (unsigned)S2, 64u, 32u, 64u,
                    (unsigned long long)S2);
    }

    for (int kc = 0; kc < S2; kc += 32) {
        const int cur = (kc >> 5) & 1;
        if (threadIdx.x < 32) __builtin_amdgcn_s_wait_tensorcnt(0);
        __syncthreads();
        if (kc + 32 < S2 && threadIdx.x < 32) {
            tdm_load_2d(lds_off(Ksm[cur ^ 1]), Kbase + (size_t)(kc + 32) * HDIM,
                        64u, (unsigned)(S2 - kc - 32), 64u, 32u, 64ull);
            tdm_load_2d(lds_off(Vsm[cur ^ 1]), Vbase + (kc + 32),
                        (unsigned)(S2 - kc - 32), 64u, 32u, 64u,
                        (unsigned long long)S2);
        }

        // ---- S = Q K^T for this 32-wide key chunk (K from LDS) ----
        Frag16 bk[2][2];
#pragma unroll
        for (int nt = 0; nt < 2; ++nt)
#pragma unroll
            for (int d = 0; d < 2; ++d) {
                const __bf16* p =
                    &Ksm[cur][(nt * 16 + l15) * 64 + d * 32 + g * 8];
                bk[nt][d].q[0] = *(const uint4*)(p);
                bk[nt][d].q[1] = *(const uint4*)(p + 16);
            }
        v8f sacc[2][2] = {};
#pragma unroll
        for (int mt = 0; mt < 2; ++mt)
#pragma unroll
            for (int nt = 0; nt < 2; ++nt) {
                sacc[mt][nt] = __builtin_amdgcn_wmma_f32_16x16x32_bf16(
                    false, aq[mt][0].v, false, bk[nt][0].v, (short)0,
                    sacc[mt][nt], false, false);
                sacc[mt][nt] = __builtin_amdgcn_wmma_f32_16x16x32_bf16(
                    false, aq[mt][1].v, false, bk[nt][1].v, (short)0,
                    sacc[mt][nt], false, false);
            }

        // ---- online softmax + P (bf16) into per-wave LDS, row-major ----
#pragma unroll
        for (int mt = 0; mt < 2; ++mt) {
#pragma unroll
            for (int v = 0; v < 8; ++v) {
                const float s0 = sacc[mt][0][v];
                const float s1 = sacc[mt][1][v];
                float mx = fmaxf(s0, s1);
                mx = fmaxf(mx, __shfl_xor(mx, 1, 32));
                mx = fmaxf(mx, __shfl_xor(mx, 2, 32));
                mx = fmaxf(mx, __shfl_xor(mx, 4, 32));
                mx = fmaxf(mx, __shfl_xor(mx, 8, 32));
                const float mnew = fmaxf(mrun[mt][v], mx);
                const float corr = __expf(mrun[mt][v] - mnew);
                mrun[mt][v] = mnew;
                const float p0 = __expf(s0 - mnew);
                const float p1 = __expf(s1 - mnew);
                float ps = p0 + p1;
                ps += __shfl_xor(ps, 1, 32);
                ps += __shfl_xor(ps, 2, 32);
                ps += __shfl_xor(ps, 4, 32);
                ps += __shfl_xor(ps, 8, 32);
                lrun[mt][v] = lrun[mt][v] * corr + ps;
#pragma unroll
                for (int nt = 0; nt < 4; ++nt) oacc[mt][nt][v] *= corr;
                const int prow = mt * 16 + v + 8 * g;
                lp[prow * 32 + l15]      = (__bf16)p0;
                lp[prow * 32 + 16 + l15] = (__bf16)p1;
            }
        }
        asm volatile("s_wait_dscnt 0" ::: "memory");  // per-wave LDS RAW

        // ---- P A-fragments from LDS, V B-fragments from LDS ----
        Frag16 pa[2];
#pragma unroll
        for (int mt = 0; mt < 2; ++mt) {
            const __bf16* p = lp + (mt * 16 + l15) * 32 + g * 8;
            pa[mt].q[0] = *(const uint4*)(p);
            pa[mt].q[1] = *(const uint4*)(p + 16);
        }
        Frag16 bv[4];
#pragma unroll
        for (int nt = 0; nt < 4; ++nt) {
            const __bf16* p = &Vsm[cur][(nt * 16 + l15) * 32 + g * 8];
            bv[nt].q[0] = *(const uint4*)(p);
            bv[nt].q[1] = *(const uint4*)(p + 16);
        }
#pragma unroll
        for (int mt = 0; mt < 2; ++mt)
#pragma unroll
            for (int nt = 0; nt < 4; ++nt)
                oacc[mt][nt] = __builtin_amdgcn_wmma_f32_16x16x32_bf16(
                    false, pa[mt].v, false, bv[nt].v, (short)0, oacc[mt][nt],
                    false, false);

        asm volatile("s_wait_dscnt 0" ::: "memory");  // WAR vs next TDM write
        __syncthreads();
    }

    // ---- normalize and write O as bf16 in (b, s1, H*hd) layout ----
#pragma unroll
    for (int mt = 0; mt < 2; ++mt) {
#pragma unroll
        for (int nt = 0; nt < 4; ++nt) {
#pragma unroll
            for (int v = 0; v < 8; ++v) {
                const float val = oacc[mt][nt][v] / lrun[mt][v];
                const int q   = q0 + mt * 16 + v + 8 * g;
                const int col = h * HDIM + nt * 16 + l15;
                Ob[(size_t)(bpos * S1 + q) * CDIM + col] = (__bf16)val;
            }
        }
    }
}

// ---------------------------------------------------------------------------
// Host-side launcher
// ---------------------------------------------------------------------------
extern "C" void kernel_launch(void* const* d_in, const int* in_sizes, int n_in,
                              void* d_out, int out_size, void* d_ws, size_t ws_size,
                              hipStream_t stream) {
    (void)in_sizes; (void)n_in; (void)out_size; (void)ws_size;

    const float* x     = (const float*)d_in[0];
    const float* y     = (const float*)d_in[1];
    const float* q_w   = (const float*)d_in[2];
    const float* q_b   = (const float*)d_in[3];
    const float* kv_w  = (const float*)d_in[4];
    const float* kv_b  = (const float*)d_in[5];
    const float* qn_w  = (const float*)d_in[6];
    const float* qn_b  = (const float*)d_in[7];
    const float* kn_w  = (const float*)d_in[8];
    const float* kn_b  = (const float*)d_in[9];
    const float* out_w = (const float*)d_in[10];
    const float* out_b = (const float*)d_in[11];
    float* out = (float*)d_out;

    char* ws = (char*)d_ws;
    const size_t OFF_XB   = 0;
    const size_t OFF_YB   = OFF_XB   + (size_t)4096 * 1024 * 2;
    const size_t OFF_QWB  = OFF_YB   + (size_t)2048 * 1024 * 2;
    const size_t OFF_KVWB = OFF_QWB  + (size_t)1024 * 1024 * 2;
    const size_t OFF_OWB  = OFF_KVWB + (size_t)2048 * 1024 * 2;
    const size_t OFF_QF   = OFF_OWB  + (size_t)1024 * 1024 * 2;   // f32 Q proj
    const size_t OFF_KVF  = OFF_QF   + (size_t)4096 * 1024 * 4;   // f32 KV proj
    const size_t OFF_QBF  = OFF_KVF  + (size_t)2048 * 2048 * 4;   // bf16 Q (b,H,s1,hd)
    const size_t OFF_KBF  = OFF_QBF  + (size_t)4096 * 1024 * 2;   // bf16 K (b,H,s2,hd)
    const size_t OFF_VTB  = OFF_KBF  + (size_t)2048 * 1024 * 2;   // bf16 V^T (b,H,hd,s2)
    const size_t OFF_OBF  = OFF_VTB  + (size_t)2048 * 1024 * 2;   // bf16 O (b,s1,C)

    __bf16* xb   = (__bf16*)(ws + OFF_XB);
    __bf16* yb   = (__bf16*)(ws + OFF_YB);
    __bf16* qwb  = (__bf16*)(ws + OFF_QWB);
    __bf16* kvwb = (__bf16*)(ws + OFF_KVWB);
    __bf16* owb  = (__bf16*)(ws + OFF_OWB);
    float*  Qf   = (float*)(ws + OFF_QF);
    float*  KVf  = (float*)(ws + OFF_KVF);
    __bf16* Qbf  = (__bf16*)(ws + OFF_QBF);
    __bf16* Kbf  = (__bf16*)(ws + OFF_KBF);
    __bf16* Vtb  = (__bf16*)(ws + OFF_VTB);
    __bf16* Obf  = (__bf16*)(ws + OFF_OBF);

    cvt_f32_bf16<<<2048, 256, 0, stream>>>(x,     xb,   4096 * 1024);
    cvt_f32_bf16<<<2048, 256, 0, stream>>>(y,     yb,   2048 * 1024);
    cvt_f32_bf16<<<2048, 256, 0, stream>>>(q_w,   qwb,  1024 * 1024);
    cvt_f32_bf16<<<2048, 256, 0, stream>>>(kv_w,  kvwb, 2048 * 1024);
    cvt_f32_bf16<<<2048, 256, 0, stream>>>(out_w, owb,  1024 * 1024);

    gemm_bf16_wmma<<<dim3(8, 32),  128, 0, stream>>>(xb, qwb,  q_b,  Qf,
                                                     4096, 1024, 1024);
    gemm_bf16_wmma<<<dim3(16, 16), 128, 0, stream>>>(yb, kvwb, kv_b, KVf,
                                                     2048, 2048, 1024);

    ln_rope_q<<<16384, 128, 0, stream>>>(Qf,  qn_w, qn_b, Qbf);
    ln_k_tr_v<<<8192,  128, 0, stream>>>(KVf, kn_w, kn_b, Kbf, Vtb);

    attention_wmma<<<dim3(16, 32), 128, 0, stream>>>(Qbf, Kbf, Vtb, Obf);

    gemm_bf16_wmma<<<dim3(8, 32), 128, 0, stream>>>(Obf, owb, out_b, out,
                                                    4096, 1024, 1024);
}
